// MemoryEfficientLocalAttention_23330262352584
// MI455X (gfx1250) — matrix operand order repaired and checked by reading, not compiled, not run on hardware
//
#include <hip/hip_runtime.h>
#include <hip/hip_bf16.h>

// ---------------------------------------------------------------------------
// MemoryEfficientLocalAttention for MI455X (gfx1250, wave32, WMMA)
//   B=2, N=4096, D_MODEL=512, H=8, D_HEAD=64, K=16, D_POS=128 (only :64 used)
// Pipeline:
//   1) knn_kernel            : brute-force kNN (async-LDS tiled, reg top-16)
//   2) qkv_gemm_kernel       : X@{Wq,Wk,Wv} via v_wmma_f32_16x16x32_f16
//   3) attn_kernel           : gather + pos-enc + softmax (one wave per n,h)
//   4) out_proj_ln_kernel    : att@Wo + bias + residual + LayerNorm (WMMA)
// Workspace usage: ~64.6 MB (idx + Q + K + V + att), carved from d_ws.
// ---------------------------------------------------------------------------

#define BATCH   2
#define NPTS    4096
#define DMODEL  512
#define NHEADS  8
#define DHEAD   64
#define KNN     16
#define MROWS   (BATCH * NPTS)      // 8192
#define LN_EPS  1e-5f

typedef __attribute__((ext_vector_type(16))) _Float16 v16h;
typedef __attribute__((ext_vector_type(8)))  float    v8f;
typedef __attribute__((ext_vector_type(4)))  float    f4;
typedef __attribute__((ext_vector_type(4)))  _Float16 h4;
typedef __attribute__((ext_vector_type(4)))  int      v4i;

// ---------------------------------------------------------------------------
// Optional CDNA5 async Global->LDS copy (ASYNCcnt-tracked). The builtin takes
// generic int4* pointers (clang inserts the AS1/AS3 addrspace casts).
// ---------------------------------------------------------------------------
#if defined(__has_builtin)
#if __has_builtin(__builtin_amdgcn_global_load_async_to_lds_b128) && \
    __has_builtin(__builtin_amdgcn_s_wait_asynccnt)
#define HAVE_ASYNC_LDS 1
#endif
#endif

#ifdef HAVE_ASYNC_LDS
__device__ inline void async_copy_b128(void* lds_dst, const void* gsrc) {
    __builtin_amdgcn_global_load_async_to_lds_b128((v4i*)gsrc, (v4i*)lds_dst, 0, 0);
}
#endif

// ---------------------------------------------------------------------------
// WMMA fragment loaders.
// LDS A tile stored row-major [m][k] (ld = K stride).
// A fragment k order used here, same as B (16x32): lanes 0-15 = M rows, elems 0..15 -> K = half*16+e
// (half = lane>>4), 16 contiguous halfs per lane.
// LDS B tile stored transposed [n][k]: lane n = lane&15, elems 0..15 ->
// K = half*16 + e, i.e. 16 contiguous halfs per lane.
// ---------------------------------------------------------------------------
__device__ inline v16h load_frag_a(const _Float16* __restrict__ s, int ld, int lane) {
    const int m    = lane & 15;
    const int half = lane >> 4;
    const _Float16* p = s + m * ld + half * 16;
    v16h a;
#pragma unroll
    for (int e = 0; e < 16; ++e) a[e] = p[e];
    return a;
}

__device__ inline v16h load_frag_b(const _Float16* __restrict__ s, int ld, int lane) {
    const int n    = lane & 15;
    const int half = lane >> 4;
    const _Float16* p = s + n * ld + half * 16;
    v16h b;
#pragma unroll
    for (int e = 0; e < 16; ++e) b[e] = p[e];
    return b;
}

// ---------------------------------------------------------------------------
// Kernel 1: brute-force kNN. One thread per query point; position tiles
// (256 pts x 12 B = 3 KB) streamed into LDS with async b128 copies when
// available. Register-resident 16-entry sorted insertion list.
// ---------------------------------------------------------------------------
__global__ __launch_bounds__(256) void knn_kernel(const float* __restrict__ pos,
                                                  int* __restrict__ idx_out) {
    const int blocksPerBatch = NPTS / 256;
    const int b = blockIdx.x / blocksPerBatch;
    const int i = (blockIdx.x % blocksPerBatch) * 256 + threadIdx.x;

    const float* pb = pos + (size_t)b * NPTS * 3;
    const float px = pb[i * 3 + 0];
    const float py = pb[i * 3 + 1];
    const float pz = pb[i * 3 + 2];

    float bd[KNN];
    int   bi[KNN];
#pragma unroll
    for (int s = 0; s < KNN; ++s) { bd[s] = 3.0e38f; bi[s] = 0; }

    __shared__ float sp[256 * 3];   // interleaved xyz, 3 KB tile

    for (int t = 0; t < NPTS; t += 256) {
        __syncthreads();
        const float* tileSrc = pb + (size_t)t * 3;
#ifdef HAVE_ASYNC_LDS
        // 3072 bytes = 192 x b128 async transfers, ASYNCcnt-tracked.
        if (threadIdx.x < 192)
            async_copy_b128(&sp[threadIdx.x * 4], tileSrc + threadIdx.x * 4);
        __builtin_amdgcn_s_wait_asynccnt(0);
#else
        for (int c = threadIdx.x; c < 256 * 3; c += 256) sp[c] = tileSrc[c];
#endif
        __syncthreads();

        if (t + 256 < NPTS) {
            // speculative prefetch of the next tile -> global_prefetch_b8
            __builtin_prefetch(pb + (size_t)(t + 256 + (int)threadIdx.x) * 3, 0, 0);
        }

        for (int jj = 0; jj < 256; ++jj) {
            const int j = t + jj;
            const float dx = px - sp[jj * 3 + 0];
            const float dy = py - sp[jj * 3 + 1];
            const float dz = pz - sp[jj * 3 + 2];
            const float d = dx * dx + dy * dy + dz * dz;
            if (j != i && d < bd[KNN - 1]) {
                float dk = d; int ik = j;
#pragma unroll
                for (int s = 0; s < KNN; ++s) {
                    if (dk < bd[s]) {
                        float td = bd[s]; bd[s] = dk; dk = td;
                        int   ti = bi[s]; bi[s] = ik; ik = ti;
                    }
                }
            }
        }
    }

    int* op = idx_out + ((size_t)b * NPTS + i) * KNN;
#pragma unroll
    for (int s = 0; s < KNN; ++s) op[s] = bi[s];
}

// ---------------------------------------------------------------------------
// Kernel 2: fused QKV projection GEMM, f16 WMMA with f32 accumulate.
// grid = (MROWS/64, DMODEL/64, 3); 256 threads = 8 waves.
// Block tile 64x64; wave w computes 16 rows x 32 cols (two 16x16x32 WMMAs
// per K-step). Staging: global_load_b128 (float4) + packed half ds_store_b64.
// Output written head-major [B,H,N,DHEAD] for contiguous gather rows.
// ---------------------------------------------------------------------------
__global__ __launch_bounds__(256) void qkv_gemm_kernel(const float* __restrict__ X,
                                                       const float* __restrict__ Wq,
                                                       const float* __restrict__ Wk,
                                                       const float* __restrict__ Wv,
                                                       float* __restrict__ Q,
                                                       float* __restrict__ Km,
                                                       float* __restrict__ Vm) {
    const int m0 = blockIdx.x * 64;
    const int n0 = blockIdx.y * 64;
    const int which = blockIdx.z;
    const float* W = (which == 0) ? Wq : (which == 1) ? Wk : Wv;
    float* O = (which == 0) ? Q : (which == 1) ? Km : Vm;

    __shared__ _Float16 sX[64][32];   // [m][k]
    __shared__ _Float16 sW[64][32];   // [n][k] (transposed weight tile)

    const int tid   = threadIdx.x;
    const int wave  = tid >> 5;
    const int lane  = tid & 31;
    const int rTile = wave & 3;       // 0..3 : 16-row tile
    const int cHalf = wave >> 2;      // 0..1 : 32-col half

    v8f acc0 = {};
    v8f acc1 = {};

    for (int kk = 0; kk < DMODEL; kk += 32) {
        __syncthreads();
        // Stage X tile: 512 float4 chunks, 2 per thread, packed-half stores.
#pragma unroll
        for (int t = tid; t < 64 * 8; t += 256) {
            const int r = t >> 3, c4 = (t & 7) * 4;
            const f4 xv = *(const f4*)&X[(size_t)(m0 + r) * DMODEL + kk + c4];
            *(h4*)&sX[r][c4] = __builtin_convertvector(xv, h4);
        }
        // Stage W tile transposed: float4 along n (coalesced), scatter to LDS.
#pragma unroll
        for (int t = tid; t < 16 * 32; t += 256) {
            const int kr = t >> 4, n4 = (t & 15) * 4;
            const f4 wv = *(const f4*)&W[(size_t)(kk + kr) * DMODEL + n0 + n4];
            sW[n4 + 0][kr] = (_Float16)wv[0];
            sW[n4 + 1][kr] = (_Float16)wv[1];
            sW[n4 + 2][kr] = (_Float16)wv[2];
            sW[n4 + 3][kr] = (_Float16)wv[3];
        }
        __syncthreads();

        if (kk + 32 < DMODEL) {
            __builtin_prefetch(&X[(size_t)(m0 + (tid >> 2)) * DMODEL + kk + 32], 0, 0);
            __builtin_prefetch(&W[(size_t)(kk + 32 + (tid & 31)) * DMODEL + n0], 0, 0);
        }

        const v16h a  = load_frag_a(&sX[rTile * 16][0], 32, lane);
        const v16h b0 = load_frag_b(&sW[cHalf * 32 + 0][0], 32, lane);
        const v16h b1 = load_frag_b(&sW[cHalf * 32 + 16][0], 32, lane);
        acc0 = __builtin_amdgcn_wmma_f32_16x16x32_f16(false, a, false, b0,
                                                      (short)0, acc0, false, false);
        acc1 = __builtin_amdgcn_wmma_f32_16x16x32_f16(false, a, false, b1,
                                                      (short)0, acc1, false, false);
    }

    // Write out: C layout VGPR r -> M = r + 8*(lane>>4), N = lane&15
    const int half = lane >> 4;
    const int nlo  = lane & 15;
#pragma unroll
    for (int r = 0; r < 8; ++r) {
        const int row = m0 + rTile * 16 + r + 8 * half;
        const int bidx = row >> 12;            // /4096
        const int nidx = row & (NPTS - 1);
#pragma unroll
        for (int ct = 0; ct < 2; ++ct) {
            const int col = n0 + cHalf * 32 + ct * 16 + nlo;
            const int h  = col >> 6;
            const int dh = col & 63;
            const float val = (ct == 0) ? acc0[r] : acc1[r];
            O[(((size_t)bidx * NHEADS + h) * NPTS + nidx) * DHEAD + dh] = val;
        }
    }
}

// ---------------------------------------------------------------------------
// Kernel 3: local attention. grid = B*N blocks, 256 threads = 8 waves,
// wave h handles head h of point n. Each lane owns 2 consecutive head dims
// -> all Q/K/V row traffic is float2 (global_load_b64), L2-resident gathers.
// ---------------------------------------------------------------------------
__global__ __launch_bounds__(256) void attn_kernel(const float* __restrict__ pos,
                                                   const float* __restrict__ Q,
                                                   const float* __restrict__ Km,
                                                   const float* __restrict__ Vm,
                                                   const int* __restrict__ idx,
                                                   const float* __restrict__ Wpos,
                                                   const float* __restrict__ bpos,
                                                   const float* __restrict__ temperature,
                                                   float* __restrict__ att) {
    const int bn = blockIdx.x;
    const int b  = bn >> 12;
    const int n  = bn & (NPTS - 1);

    __shared__ int   sidx[KNN];
    __shared__ float srel[KNN][3];
    __shared__ float sWp[3][DHEAD];
    __shared__ float sbp[DHEAD];

    const int tid = threadIdx.x;
    if (tid < KNN) sidx[tid] = idx[((size_t)b * NPTS + n) * KNN + tid];
    if (tid < 3 * DHEAD) {
        const int c = tid / DHEAD, d = tid % DHEAD;
        sWp[c][d] = Wpos[c * 128 + d];        // only the first 64 of D_POS matter
    }
    if (tid >= 192 && tid < 192 + DHEAD) sbp[tid - 192] = bpos[tid - 192];
    __syncthreads();
    if (tid < KNN * 3) {
        const int k = tid / 3, c = tid % 3;
        const int j = sidx[k];
        srel[k][c] = pos[((size_t)b * NPTS + n) * 3 + c] -
                     pos[((size_t)b * NPTS + j) * 3 + c];
    }
    __syncthreads();

    const int h    = tid >> 5;
    const int lane = tid & 31;
    const int d0   = lane * 2;

    const size_t headBase = ((size_t)b * NHEADS + h) * NPTS;
    const float2 qv = *(const float2*)&Q[(headBase + n) * DHEAD + d0];

    float sc[KNN];
#pragma unroll
    for (int k = 0; k < KNN; ++k) {
        const int j = sidx[k];
        const float2 kv = *(const float2*)&Km[(headBase + j) * DHEAD + d0];
        const float pe0 = sbp[d0] + srel[k][0] * sWp[0][d0] +
                          srel[k][1] * sWp[1][d0] + srel[k][2] * sWp[2][d0];
        const float pe1 = sbp[d0 + 1] + srel[k][0] * sWp[0][d0 + 1] +
                          srel[k][1] * sWp[1][d0 + 1] + srel[k][2] * sWp[2][d0 + 1];
        float p = qv.x * (kv.x + pe0) + qv.y * (kv.y + pe1);
#pragma unroll
        for (int s = 16; s >= 1; s >>= 1) p += __shfl_xor(p, s, 32);
        sc[k] = p;
    }

    const float tinv = 1.0f / temperature[0];
    float m = sc[0] * tinv;
#pragma unroll
    for (int k = 1; k < KNN; ++k) m = fmaxf(m, sc[k] * tinv);
    float sum = 0.0f;
#pragma unroll
    for (int k = 0; k < KNN; ++k) {
        sc[k] = __expf(sc[k] * tinv - m);
        sum += sc[k];
    }
    const float rs = 1.0f / sum;

    float acc0 = 0.0f, acc1 = 0.0f;
#pragma unroll
    for (int k = 0; k < KNN; ++k) {
        const int j = sidx[k];
        const float2 vv = *(const float2*)&Vm[(headBase + j) * DHEAD + d0];
        const float w = sc[k] * rs;
        acc0 += w * vv.x;
        acc1 += w * vv.y;
    }

    float2* orow = (float2*)&att[(size_t)bn * DMODEL + h * DHEAD + d0];
    *orow = make_float2(acc0, acc1);
}

// ---------------------------------------------------------------------------
// Kernel 4: output projection + bias + residual + LayerNorm, fused.
// grid = MROWS/16 blocks, 256 threads. Each block computes a 16x512 output
// strip: 8 waves, wave w owns a 64-col slice (4 WMMA accumulators), K-loop
// over the shared 16x32 A tile. Then the same block does the LayerNorm.
// ---------------------------------------------------------------------------
__global__ __launch_bounds__(256) void out_proj_ln_kernel(const float* __restrict__ att,
                                                          const float* __restrict__ Wo,
                                                          const float* __restrict__ bo,
                                                          const float* __restrict__ feats,
                                                          const float* __restrict__ gamma,
                                                          const float* __restrict__ beta,
                                                          float* __restrict__ out) {
    const int m0 = blockIdx.x * 16;

    __shared__ _Float16 sA[16][32];        // att tile [m][k]
    __shared__ _Float16 sB[DMODEL][32];    // Wo tile transposed [n][k]  (32 KB)
    __shared__ float    sOut[16][DMODEL];  // projected strip            (32 KB)

    const int tid  = threadIdx.x;
    const int wave = tid >> 5;
    const int lane = tid & 31;

    v8f acc[4] = {{}, {}, {}, {}};

    for (int kk = 0; kk < DMODEL; kk += 32) {
        __syncthreads();
        // A tile: 128 float4 chunks.
        if (tid < 128) {
            const int r = tid >> 3, c4 = (tid & 7) * 4;
            const f4 av = *(const f4*)&att[(size_t)(m0 + r) * DMODEL + kk + c4];
            *(h4*)&sA[r][c4] = __builtin_convertvector(av, h4);
        }
        // Wo tile transposed: 4096 float4 chunks along n, 16 per thread.
#pragma unroll
        for (int t = tid; t < 32 * (DMODEL / 4); t += 256) {
            const int kr = t >> 7, n4 = (t & 127) * 4;
            const f4 wv = *(const f4*)&Wo[(size_t)(kk + kr) * DMODEL + n4];
            sB[n4 + 0][kr] = (_Float16)wv[0];
            sB[n4 + 1][kr] = (_Float16)wv[1];
            sB[n4 + 2][kr] = (_Float16)wv[2];
            sB[n4 + 3][kr] = (_Float16)wv[3];
        }
        __syncthreads();

        const v16h a = load_frag_a(&sA[0][0], 32, lane);
#pragma unroll
        for (int ct = 0; ct < 4; ++ct) {
            const v16h bf = load_frag_b(&sB[wave * 64 + ct * 16][0], 32, lane);
            acc[ct] = __builtin_amdgcn_wmma_f32_16x16x32_f16(false, a, false, bf,
                                                             (short)0, acc[ct],
                                                             false, false);
        }
    }

    // Spill accumulators to LDS: M = r + 8*(lane>>4), N = lane&15
    const int half = lane >> 4;
    const int nlo  = lane & 15;
#pragma unroll
    for (int ct = 0; ct < 4; ++ct) {
#pragma unroll
        for (int r = 0; r < 8; ++r) {
            sOut[r + 8 * half][wave * 64 + ct * 16 + nlo] = acc[ct][r];
        }
    }
    __syncthreads();

    // Fused bias + residual + LayerNorm. Wave w handles rows 2w and 2w+1;
    // each lane owns 16 contiguous columns.
#pragma unroll
    for (int rr = 0; rr < 2; ++rr) {
        const int mloc = wave * 2 + rr;
        const int row  = m0 + mloc;
        float x[16];
        float sum = 0.0f, sq = 0.0f;
#pragma unroll
        for (int c = 0; c < 16; ++c) {
            const int col = lane * 16 + c;
            const float v = sOut[mloc][col] + bo[col] +
                            feats[(size_t)row * DMODEL + col];
            x[c] = v;
            sum += v;
            sq  += v * v;
        }
#pragma unroll
        for (int s = 16; s >= 1; s >>= 1) {
            sum += __shfl_xor(sum, s, 32);
            sq  += __shfl_xor(sq,  s, 32);
        }
        const float mu   = sum * (1.0f / DMODEL);
        const float var  = sq * (1.0f / DMODEL) - mu * mu;
        const float rstd = rsqrtf(var + LN_EPS);
#pragma unroll
        for (int c = 0; c < 16; ++c) {
            const int col = lane * 16 + c;
            out[(size_t)row * DMODEL + col] =
                (x[c] - mu) * rstd * gamma[col] + beta[col];
        }
    }
}

// ---------------------------------------------------------------------------
// Host launcher
// ---------------------------------------------------------------------------
extern "C" void kernel_launch(void* const* d_in, const int* in_sizes, int n_in,
                              void* d_out, int out_size, void* d_ws, size_t ws_size,
                              hipStream_t stream) {
    const float* positions   = (const float*)d_in[0];
    const float* features    = (const float*)d_in[1];
    // d_in[2] = k_neighbors (compile-time KNN=16)
    const float* Wq          = (const float*)d_in[3];
    const float* Wk          = (const float*)d_in[4];
    const float* Wv          = (const float*)d_in[5];
    const float* Wo          = (const float*)d_in[6];
    const float* bo          = (const float*)d_in[7];
    const float* Wpos        = (const float*)d_in[8];
    const float* bpos        = (const float*)d_in[9];
    const float* temperature = (const float*)d_in[10];
    const float* ln_gamma    = (const float*)d_in[11];
    const float* ln_beta     = (const float*)d_in[12];

    // Workspace carve-out (~64.6 MB)
    char* p = (char*)d_ws;
    int*   idx = (int*)p;    p += (size_t)MROWS * KNN * sizeof(int);      // 512 KB
    float* q   = (float*)p;  p += (size_t)MROWS * DMODEL * sizeof(float); // 16 MB
    float* k   = (float*)p;  p += (size_t)MROWS * DMODEL * sizeof(float); // 16 MB
    float* v   = (float*)p;  p += (size_t)MROWS * DMODEL * sizeof(float); // 16 MB
    float* att = (float*)p;                                               // 16 MB

    // 1) kNN
    knn_kernel<<<BATCH * (NPTS / 256), 256, 0, stream>>>(positions, idx);

    // 2) Fused QKV projections (WMMA)
    qkv_gemm_kernel<<<dim3(MROWS / 64, DMODEL / 64, 3), 256, 0, stream>>>(
        features, Wq, Wk, Wv, q, k, v);

    // 3) Local attention with positional encoding + softmax
    attn_kernel<<<BATCH * NPTS, 256, 0, stream>>>(
        positions, q, k, v, idx, Wpos, bpos, temperature, att);

    // 4) Output projection + residual + LayerNorm (WMMA)
    out_proj_ln_kernel<<<MROWS / 16, 256, 0, stream>>>(
        att, Wo, bo, features, ln_gamma, ln_beta, (float*)d_out);
}
